// Model_78477642433020
// MI455X (gfx1250) — matrix operand order repaired and checked by reading, not compiled, not run on hardware
//
#include <hip/hip_runtime.h>
#include <hip/hip_bf16.h>

#define SEQ   2048
#define BATCH 4
#define EMB   768
#define NH    12
#define HD    64
#define FFD   3072
#define MTOK  (SEQ*BATCH)   // 8192 tokens

typedef __attribute__((ext_vector_type(16))) _Float16 v16h;
typedef __attribute__((ext_vector_type(8)))  float    v8f;

union V16 { v16h v; uint4 q[2]; };

__device__ __forceinline__ v8f wmma16x16x32(V16 a, V16 b, v8f c) {
  // D = A(16x32 f16) * B(32x16 f16) + C(16x16 f32)
  return __builtin_amdgcn_wmma_f32_16x16x32_f16(false, a.v, false, b.v,
                                                (short)0, c, false, false);
}

// ---------------------------------------------------------------- f32 -> f16
// Vectorized: b128 load -> b64 store. All array sizes are multiples of 4.
struct __align__(8) h4 { _Float16 x, y, z, w; };

__global__ void cvt_f32_f16_v4(const float4* __restrict__ in,
                               h4* __restrict__ out, int n4) {
  int i = blockIdx.x * blockDim.x + threadIdx.x;
  if (i < n4) {
    const float4 v = in[i];
    h4 o;
    o.x = (_Float16)v.x; o.y = (_Float16)v.y;
    o.z = (_Float16)v.z; o.w = (_Float16)v.w;
    out[i] = o;
  }
}

// ------------------------------------------------------- WMMA GEMM: A * W^T
// A: M x K (f16 row-major), W: N x K (f16 row-major, i.e. x @ W.T)
// Each wave computes a 32x64 tile (2 M-subtiles x 4 N-subtiles) so every B
// fragment feeds 2 WMMAs. Block = 256 threads = 8 waves = 256 rows.
template<typename OutT, bool RELU, bool RES>
__global__ __launch_bounds__(256) void gemm_wmma(
    const _Float16* __restrict__ A, const _Float16* __restrict__ W,
    const float* __restrict__ bias, const float* __restrict__ resid,
    OutT* __restrict__ out, int M, int N, int K)
{
  const int lane = threadIdx.x & 31;
  const int wave = threadIdx.x >> 5;
  const int half = lane >> 4;
  const int l15  = lane & 15;
  const int m0 = (blockIdx.y * 8 + wave) * 32;
  const int n0 = blockIdx.x * 64;

  const _Float16* arow0 = A + (size_t)(m0 + l15) * K;            // A subtile 0
  const _Float16* arow1 = A + (size_t)(m0 + 16 + l15) * K;       // A subtile 1
  const _Float16* brow  = W + (size_t)(n0 + l15) * K + 8 * half;
  const size_t bts = (size_t)16 * K;

  v8f acc[2][4] = {};
  for (int kb = 0; kb < K; kb += 32) {
    V16 a0, a1;
    a0.q[0] = *(const uint4*)(arow0 + kb + 8 * half);      // K = 8h .. 8h+7
    a0.q[1] = *(const uint4*)(arow0 + kb + 16 + 8 * half); // K = 16+8h ..
    a1.q[0] = *(const uint4*)(arow1 + kb + 8 * half);
    a1.q[1] = *(const uint4*)(arow1 + kb + 16 + 8 * half);
#pragma unroll
    for (int t = 0; t < 4; ++t) {
      V16 b;
      const _Float16* bp = brow + (size_t)t * bts + kb;
      b.q[0] = *(const uint4*)(bp);
      b.q[1] = *(const uint4*)(bp + 16);
      acc[0][t] = wmma16x16x32(a0, b, acc[0][t]);
      acc[1][t] = wmma16x16x32(a1, b, acc[1][t]);
    }
  }
#pragma unroll
  for (int u = 0; u < 2; ++u) {
#pragma unroll
    for (int t = 0; t < 4; ++t) {
#pragma unroll
      for (int r = 0; r < 8; ++r) {
        const int m = m0 + u * 16 + r + 8 * half;  // C layout: row = r + 8*half
        const int n = n0 + t * 16 + l15;           //           col = lane & 15
        float v = acc[u][t][r] + bias[n];
        if constexpr (RES)  v += resid[(size_t)m * N + n];
        if constexpr (RELU) v = fmaxf(v, 0.0f);
        out[(size_t)m * N + n] = (OutT)v;
      }
    }
  }
}

// ---------------------------------------------- split QKV + RoPE + V transpose
// proj: (S*B) x 3E f16. Writes Q,K (b,h,s,d) rotated; V^T (b,h,d,s).
__global__ void rope_qkv(const _Float16* __restrict__ proj,
                         const float* __restrict__ sinT, const float* __restrict__ cosT,
                         _Float16* __restrict__ Q, _Float16* __restrict__ Kk,
                         _Float16* __restrict__ VT)
{
  const int idx = blockIdx.x * blockDim.x + threadIdx.x;
  const int i  = idx & 31;               // pair index, HD/2 = 32
  const int s  = (idx >> 5) & (SEQ - 1);
  const int bh = idx >> 16;
  if (bh >= BATCH * NH) return;
  const int b = bh / NH;
  const int h = bh % NH;
  const _Float16* row = proj + ((size_t)s * BATCH + b) * (3 * EMB) + h * HD;
  const float q0 = (float)row[2*i],         q1 = (float)row[2*i + 1];
  const float k0 = (float)row[EMB + 2*i],   k1 = (float)row[EMB + 2*i + 1];
  const float v0 = (float)row[2*EMB + 2*i], v1 = (float)row[2*EMB + 2*i + 1];
  const float sn = sinT[s * 32 + i];
  const float cs = cosT[s * 32 + i];
  const size_t basep = (size_t)bh * SEQ * HD;
  Q[basep + (size_t)s * HD + 2*i]     = (_Float16)(q0 * cs - q1 * sn);
  Q[basep + (size_t)s * HD + 2*i + 1] = (_Float16)(q1 * cs + q0 * sn);
  Kk[basep + (size_t)s * HD + 2*i]     = (_Float16)(k0 * cs - k1 * sn);
  Kk[basep + (size_t)s * HD + 2*i + 1] = (_Float16)(k1 * cs + k0 * sn);
  VT[basep + (size_t)(2*i) * SEQ + s]     = (_Float16)v0;
  VT[basep + (size_t)(2*i + 1) * SEQ + s] = (_Float16)v1;
}

// ----------------------------------------------------- causal flash attention
// One wave per (b,h, 16-query tile). Single-wave workgroup so barriers are
// pure counter-waits (no cross-wave divergence). O stored as (S,B,E) f16.
__global__ __launch_bounds__(32) void attention_fa(
    const _Float16* __restrict__ Q, const _Float16* __restrict__ Kk,
    const _Float16* __restrict__ VT, _Float16* __restrict__ O)
{
  __shared__ __align__(16) _Float16 P[16 * 32];   // C-layout -> A-layout stage
  const int lane = threadIdx.x & 31;
  const int half = lane >> 4;
  const int l15  = lane & 15;
  const int bh = blockIdx.x;
  const int b = bh / NH;
  const int h = bh % NH;
  const int q0 = blockIdx.y * 16;
  const size_t base = (size_t)bh * SEQ * HD;

  const _Float16* qrow = Q + base + (size_t)(q0 + l15) * HD;
  V16 qa0, qa1;                               // Q A-fragments, d=0..31, 32..63
  qa0.q[0] = *(const uint4*)(qrow + 16 * half);
  qa0.q[1] = *(const uint4*)(qrow + 8 + 16 * half);
  qa1.q[0] = *(const uint4*)(qrow + 32 + 16 * half);
  qa1.q[1] = *(const uint4*)(qrow + 40 + 16 * half);

  v8f o0 = {}, o1 = {}, o2 = {}, o3 = {};     // 16x64 output accumulator
  float mrun[8], lrun[8];
#pragma unroll
  for (int r = 0; r < 8; ++r) { mrun[r] = -3.0e38f; lrun[r] = 0.0f; }

  const int nblk = (q0 + 47) >> 5;            // key blocks of 32, kb <= q0+15
  for (int blk = 0; blk < nblk; ++blk) {
    const int kb = blk * 32;
    v8f s0 = {}, s1 = {};                     // scores: keys kb..+15, kb+16..+31
    {
      const _Float16* kr = Kk + base + (size_t)(kb + l15) * HD + 16 * half;
      V16 bb;
      bb.q[0] = *(const uint4*)(kr);      bb.q[1] = *(const uint4*)(kr + 8);
      s0 = wmma16x16x32(qa0, bb, s0);
      bb.q[0] = *(const uint4*)(kr + 32); bb.q[1] = *(const uint4*)(kr + 40);
      s0 = wmma16x16x32(qa1, bb, s0);
      const _Float16* kr2 = kr + (size_t)16 * HD;
      bb.q[0] = *(const uint4*)(kr2);      bb.q[1] = *(const uint4*)(kr2 + 8);
      s1 = wmma16x16x32(qa0, bb, s1);
      bb.q[0] = *(const uint4*)(kr2 + 32); bb.q[1] = *(const uint4*)(kr2 + 40);
      s1 = wmma16x16x32(qa1, bb, s1);
    }
    const float sc = 0.125f;                  // 1/sqrt(64)
    float p0v[8], p1v[8];
#pragma unroll
    for (int r = 0; r < 8; ++r) {
      const int m = q0 + r + 8 * half;
      float a = s0[r] * sc; if (kb + l15 > m)      a = -3.0e38f;  // causal
      float c = s1[r] * sc; if (kb + 16 + l15 > m) c = -3.0e38f;
      float rv = fmaxf(a, c);                 // row max across 16 lanes of half
      rv = fmaxf(rv, __shfl_xor(rv, 1));
      rv = fmaxf(rv, __shfl_xor(rv, 2));
      rv = fmaxf(rv, __shfl_xor(rv, 4));
      rv = fmaxf(rv, __shfl_xor(rv, 8));
      const float nm = fmaxf(mrun[r], rv);
      const float al = __expf(mrun[r] - nm);
      const float p0 = __expf(a - nm);
      const float p1 = __expf(c - nm);
      float rs = p0 + p1;
      rs += __shfl_xor(rs, 1);
      rs += __shfl_xor(rs, 2);
      rs += __shfl_xor(rs, 4);
      rs += __shfl_xor(rs, 8);
      lrun[r] = lrun[r] * al + rs;
      mrun[r] = nm;
      o0[r] *= al; o1[r] *= al; o2[r] *= al; o3[r] *= al;
      p0v[r] = p0; p1v[r] = p1;
    }
#pragma unroll
    for (int r = 0; r < 8; ++r) {             // C-layout stores to LDS
      P[(r + 8 * half) * 32 + l15]      = (_Float16)p0v[r];
      P[(r + 8 * half) * 32 + 16 + l15] = (_Float16)p1v[r];
    }
    __syncthreads();                          // dscnt wait (single-wave WG)
    V16 pa;                                   // re-read in A-layout
    const _Float16* prow = P + l15 * 32;
    pa.q[0] = *(const uint4*)(prow + 16 * half);
    pa.q[1] = *(const uint4*)(prow + 8 + 16 * half);
    __syncthreads();

    const _Float16* vr = VT + base + (size_t)l15 * SEQ + kb + 16 * half;
    V16 vb;
    vb.q[0] = *(const uint4*)(vr);     vb.q[1] = *(const uint4*)(vr + 8);
    o0 = wmma16x16x32(pa, vb, o0);
    const _Float16* vr1 = vr + (size_t)16 * SEQ;
    vb.q[0] = *(const uint4*)(vr1);    vb.q[1] = *(const uint4*)(vr1 + 8);
    o1 = wmma16x16x32(pa, vb, o1);
    const _Float16* vr2 = vr + (size_t)32 * SEQ;
    vb.q[0] = *(const uint4*)(vr2);    vb.q[1] = *(const uint4*)(vr2 + 8);
    o2 = wmma16x16x32(pa, vb, o2);
    const _Float16* vr3 = vr + (size_t)48 * SEQ;
    vb.q[0] = *(const uint4*)(vr3);    vb.q[1] = *(const uint4*)(vr3 + 8);
    o3 = wmma16x16x32(pa, vb, o3);
  }

#pragma unroll
  for (int r = 0; r < 8; ++r) {
    const int m = q0 + r + 8 * half;
    const float inv = 1.0f / lrun[r];
    const size_t orow = ((size_t)m * BATCH + b) * EMB + (size_t)h * HD;
    O[orow + l15]      = (_Float16)(o0[r] * inv);
    O[orow + 16 + l15] = (_Float16)(o1[r] * inv);
    O[orow + 32 + l15] = (_Float16)(o2[r] * inv);
    O[orow + 48 + l15] = (_Float16)(o3[r] * inv);
  }
}

// -------------------------------------------------------------- LayerNorm
__global__ __launch_bounds__(256) void layernorm_kernel(
    const float* __restrict__ x, const float* __restrict__ g,
    const float* __restrict__ bt, float* __restrict__ outf,
    _Float16* __restrict__ outh, int E)
{
  __shared__ float red0[256];
  __shared__ float red1[256];
  const int row = blockIdx.x;
  const int tid = threadIdx.x;
  const float* xr = x + (size_t)row * E;
  float s = 0.f, ss = 0.f;
  for (int i = tid; i < E; i += 256) { const float v = xr[i]; s += v; ss += v * v; }
  red0[tid] = s; red1[tid] = ss;
  __syncthreads();
  for (int off = 128; off > 0; off >>= 1) {
    if (tid < off) { red0[tid] += red0[tid + off]; red1[tid] += red1[tid + off]; }
    __syncthreads();
  }
  const float mu  = red0[0] / (float)E;
  const float var = red1[0] / (float)E - mu * mu;
  const float inv = rsqrtf(var + 1e-5f);
  for (int i = tid; i < E; i += 256) {
    const float v = (xr[i] - mu) * inv * g[i] + bt[i];
    if (outf) outf[(size_t)row * E + i] = v;
    if (outh) outh[(size_t)row * E + i] = (_Float16)v;
  }
}

// ------------------------------------------------------------------ launch
extern "C" void kernel_launch(void* const* d_in, const int* in_sizes, int n_in,
                              void* d_out, int out_size, void* d_ws, size_t ws_size,
                              hipStream_t stream)
{
  (void)in_sizes; (void)n_in; (void)out_size; (void)ws_size;
  const float* x    = (const float*)d_in[0];
  // d_in[1] = attn_mask (zeros, causal applied in-kernel) -> unused
  const float* sinT = (const float*)d_in[2];
  const float* cosT = (const float*)d_in[3];
  const float* Win  = (const float*)d_in[4];
  const float* bin  = (const float*)d_in[5];
  const float* Wo   = (const float*)d_in[6];
  const float* bo   = (const float*)d_in[7];
  const float* W1   = (const float*)d_in[8];
  const float* b1   = (const float*)d_in[9];
  const float* W2   = (const float*)d_in[10];
  const float* b2   = (const float*)d_in[11];
  const float* g1   = (const float*)d_in[12];
  const float* be1  = (const float*)d_in[13];
  const float* g2   = (const float*)d_in[14];
  const float* be2  = (const float*)d_in[15];
  float* out = (float*)d_out;

  char* base = (char*)d_ws;
  size_t off = 0;
  auto take = [&](size_t elems, size_t esz) -> void* {
    void* p = base + off;
    off = (off + elems * esz + 255) & ~((size_t)255);
    return p;
  };
  _Float16* x16    = (_Float16*)take((size_t)MTOK * EMB, 2);
  _Float16* Win16  = (_Float16*)take((size_t)3 * EMB * EMB, 2);
  _Float16* Wo16   = (_Float16*)take((size_t)EMB * EMB, 2);
  _Float16* W116   = (_Float16*)take((size_t)FFD * EMB, 2);
  _Float16* W216   = (_Float16*)take((size_t)EMB * FFD, 2);
  _Float16* proj16 = (_Float16*)take((size_t)MTOK * 3 * EMB, 2);
  _Float16* Q16    = (_Float16*)take((size_t)BATCH * NH * SEQ * HD, 2);
  _Float16* K16    = (_Float16*)take((size_t)BATCH * NH * SEQ * HD, 2);
  _Float16* VT16   = (_Float16*)take((size_t)BATCH * NH * SEQ * HD, 2);
  _Float16* O16    = (_Float16*)take((size_t)MTOK * EMB, 2);
  float*    ares   = (float*)take((size_t)MTOK * EMB, 4);
  float*    y1f    = (float*)take((size_t)MTOK * EMB, 4);
  _Float16* y1h    = (_Float16*)take((size_t)MTOK * EMB, 2);
  _Float16* h1     = (_Float16*)take((size_t)MTOK * FFD, 2);
  float*    ffres  = (float*)take((size_t)MTOK * EMB, 4);

  auto cvt = [&](const float* src, _Float16* dst, size_t n) {
    const size_t n4 = n / 4;
    cvt_f32_f16_v4<<<dim3((unsigned)((n4 + 255) / 256)), dim3(256), 0, stream>>>(
        (const float4*)src, (h4*)dst, (int)n4);
  };
  cvt(x,   x16,   (size_t)MTOK * EMB);
  cvt(Win, Win16, (size_t)3 * EMB * EMB);
  cvt(Wo,  Wo16,  (size_t)EMB * EMB);
  cvt(W1,  W116,  (size_t)FFD * EMB);
  cvt(W2,  W216,  (size_t)EMB * FFD);

  // in_proj: (M x 768) @ (2304 x 768)^T -> f16 (M x 2304)
  gemm_wmma<_Float16, false, false><<<dim3(3 * EMB / 64, MTOK / 256), 256, 0, stream>>>(
      x16, Win16, bin, nullptr, proj16, MTOK, 3 * EMB, EMB);

  rope_qkv<<<dim3((BATCH * NH * SEQ * (HD / 2)) / 256), 256, 0, stream>>>(
      proj16, sinT, cosT, Q16, K16, VT16);

  attention_fa<<<dim3(BATCH * NH, SEQ / 16), 32, 0, stream>>>(Q16, K16, VT16, O16);

  // out_proj + residual(x) -> f32
  gemm_wmma<float, false, true><<<dim3(EMB / 64, MTOK / 256), 256, 0, stream>>>(
      O16, Wo16, bo, x, ares, MTOK, EMB, EMB);

  layernorm_kernel<<<dim3(MTOK), 256, 0, stream>>>(ares, g1, be1, y1f, y1h, EMB);

  // linear1 + ReLU -> f16 (M x 3072)
  gemm_wmma<_Float16, true, false><<<dim3(FFD / 64, MTOK / 256), 256, 0, stream>>>(
      y1h, W116, b1, nullptr, h1, MTOK, FFD, EMB);

  // linear2 + residual(y1) -> f32
  gemm_wmma<float, false, true><<<dim3(EMB / 64, MTOK / 256), 256, 0, stream>>>(
      h1, W216, b2, y1f, ffres, MTOK, EMB, FFD);

  layernorm_kernel<<<dim3(MTOK), 256, 0, stream>>>(ffres, g2, be2, out, nullptr, EMB);
}
